// Offset_Learning_Mask_35871566856394
// MI455X (gfx1250) — compile-verified, matrix operand order checked by reading, not run on hardware
//
#include <hip/hip_runtime.h>
#include <hip/hip_bf16.h>

typedef __attribute__((ext_vector_type(16))) __bf16 v16bf;
typedef __attribute__((ext_vector_type(8)))  float  v8f;

#define HW 16384
#define C  256
#define NQ 150
#define NP 160
#define NB 8

__device__ __forceinline__ __bf16 f2bf(float f) {
  unsigned u = __builtin_bit_cast(unsigned, f);
  unsigned short s = (unsigned short)((u + 0x7FFFu + ((u >> 16) & 1u)) >> 16);
  return __builtin_bit_cast(__bf16, s);
}

// Fragment loader for wave32 WMMA bf16 A/B operands.
// A 16x32 (M=16,K=32): lane 0-15 -> row M=lane, K 0..15 ; lane 16-31 -> row M=lane-16, K 16..31
// B 32x16 (K=32,N=16): lane 0-15 -> col N=lane, K 0..15 ; lane 16-31 -> col N=lane-16, K 16..31
// Both read 32 contiguous bytes per lane from a row-major [row][K] buffer.
__device__ __forceinline__ v16bf ld_frag(const __bf16* base, int ld, int row0, int k0, int lane) {
  const __bf16* p = base + (size_t)(row0 + (lane & 15)) * (size_t)ld + (k0 + ((lane >> 4) << 4));
  return *(const v16bf*)p;
}

__device__ __forceinline__ v8f wmma_bf16(v16bf a, v16bf b, v8f c) {
  return __builtin_amdgcn_wmma_f32_16x16x32_bf16(false, a, false, b, (short)0, c, false, false);
}

// ---------------- prep: image features -> bf16 in both layouts ----------------
__global__ void k_prep_img(const float* __restrict__ mf,
                           __bf16* __restrict__ imgT,   // [b][hw][c]
                           __bf16* __restrict__ imgC) { // [b][c][hw]
  __shared__ float tile[32][33];
  int blk = blockIdx.x;
  int pblk = blk % (HW / 32);
  int cb   = (blk / (HW / 32)) % (C / 32);
  int b    = blk / ((HW / 32) * (C / 32));
  int p0 = pblk * 32, c0 = cb * 32;
  int tx = threadIdx.x & 31, ty = threadIdx.x >> 5;
  #pragma unroll
  for (int r = 0; r < 4; ++r) {
    int ch = c0 + ty + r * 8;
    size_t idx = ((size_t)(b * C + ch)) * HW + p0 + tx;
    float f = mf[idx];
    imgC[idx] = f2bf(f);
    tile[ty + r * 8][tx] = f;
  }
  __syncthreads();
  #pragma unroll
  for (int r = 0; r < 4; ++r) {
    int p = p0 + ty + r * 8;
    imgT[((size_t)b * HW + p) * C + c0 + tx] = f2bf(tile[tx][ty + r * 8]);
  }
}

// ---------------- prep: cls (padded), clsT, weights -> bf16 ----------------
__global__ void k_prep_small(const float* __restrict__ me, const float* __restrict__ Wc,
                             const float* __restrict__ Wf, __bf16* __restrict__ cls,
                             __bf16* __restrict__ clsT, __bf16* __restrict__ wc,
                             __bf16* __restrict__ wf) {
  int i = blockIdx.x * blockDim.x + threadIdx.x;
  const int nCls = NB * NP * C;
  if (i < nCls) {
    int ch = i % C; int n = (i / C) % NP; int b = i / (C * NP);
    float v = (n < NQ) ? me[((size_t)(b * NQ + n)) * C + ch] : 0.f;
    cls[i] = f2bf(v);
    clsT[((size_t)b * C + ch) * NP + n] = f2bf(v);
  } else {
    int j = i - nCls;
    if (j < C * C) { wc[j] = f2bf(Wc[j]); wf[j] = f2bf(Wf[j]); }
  }
}

// ---------------- row (class-direction) softmax stats over hw: two-pass ----------------
__global__ __launch_bounds__(256, 1) void k_row_stats(
    const __bf16* __restrict__ cls, const __bf16* __restrict__ imgT,
    float* __restrict__ rmax, float* __restrict__ rsum) {
  int b  = blockIdx.x / (NP / 16);
  int n0 = (blockIdx.x % (NP / 16)) * 16;
  int wave = threadIdx.x >> 5, lane = threadIdx.x & 31, half = lane >> 4;
  const __bf16* clsb = cls + (size_t)b * NP * C;
  const __bf16* img  = imgT + (size_t)b * HW * C;
  v16bf afr[8];
  #pragma unroll
  for (int k = 0; k < 8; ++k) afr[k] = ld_frag(clsb, C, n0, k * 32, lane);

  __shared__ float smw[8][16], ssw[8][16], sM[16];

  // ---- pass 1: per-lane running max (no shuffles, no branches in the loop) ----
  float mx[8];
  #pragma unroll
  for (int v = 0; v < 8; ++v) mx[v] = -3.0e38f;
  for (int pt = wave; pt < HW / 16; pt += 8) {
    v8f acc = {};
    #pragma unroll
    for (int k = 0; k < 8; ++k)
      acc = wmma_bf16(afr[k], ld_frag(img, C, pt * 16, k * 32, lane), acc);
    #pragma unroll
    for (int v = 0; v < 8; ++v) mx[v] = fmaxf(mx[v], acc[v]);
  }
  #pragma unroll
  for (int v = 0; v < 8; ++v) {
    float r = mx[v];
    r = fmaxf(r, __shfl_xor(r, 1, 32));
    r = fmaxf(r, __shfl_xor(r, 2, 32));
    r = fmaxf(r, __shfl_xor(r, 4, 32));
    r = fmaxf(r, __shfl_xor(r, 8, 32));
    mx[v] = r;
  }
  if ((lane & 15) == 0) {
    #pragma unroll
    for (int v = 0; v < 8; ++v) smw[wave][v + 8 * half] = mx[v];
  }
  __syncthreads();
  if (threadIdx.x < 16) {
    float M = -3.0e38f;
    for (int w = 0; w < 8; ++w) M = fmaxf(M, smw[w][threadIdx.x]);
    sM[threadIdx.x] = M;
  }
  __syncthreads();
  float Mrow[8];
  #pragma unroll
  for (int v = 0; v < 8; ++v) Mrow[v] = sM[v + 8 * half];

  // ---- pass 2: per-lane sum of exp(x - M) (no shuffles in the loop) ----
  float sa[8];
  #pragma unroll
  for (int v = 0; v < 8; ++v) sa[v] = 0.f;
  for (int pt = wave; pt < HW / 16; pt += 8) {
    v8f acc = {};
    #pragma unroll
    for (int k = 0; k < 8; ++k)
      acc = wmma_bf16(afr[k], ld_frag(img, C, pt * 16, k * 32, lane), acc);
    #pragma unroll
    for (int v = 0; v < 8; ++v) sa[v] += __expf(acc[v] - Mrow[v]);
  }
  #pragma unroll
  for (int v = 0; v < 8; ++v) {
    float s = sa[v];
    s += __shfl_xor(s, 1, 32);
    s += __shfl_xor(s, 2, 32);
    s += __shfl_xor(s, 4, 32);
    s += __shfl_xor(s, 8, 32);
    sa[v] = s;
  }
  if ((lane & 15) == 0) {
    #pragma unroll
    for (int v = 0; v < 8; ++v) ssw[wave][v + 8 * half] = sa[v];
  }
  __syncthreads();
  if (threadIdx.x < 16) {
    float S = 0.f;
    for (int w = 0; w < 8; ++w) S += ssw[w][threadIdx.x];
    rmax[b * NP + n0 + threadIdx.x] = sM[threadIdx.x];
    rsum[b * NP + n0 + threadIdx.x] = S;
  }
}

// ---------------- column (pixel-direction) softmax stats over N ----------------
__global__ __launch_bounds__(256, 1) void k_col_stats(
    const __bf16* __restrict__ cls, const __bf16* __restrict__ imgT,
    float* __restrict__ cmax, float* __restrict__ csum) {
  int b  = blockIdx.x / (HW / 128);
  int pb = blockIdx.x % (HW / 128);
  int wave = threadIdx.x >> 5, lane = threadIdx.x & 31, half = lane >> 4;
  int p0 = pb * 128 + wave * 16;
  const __bf16* clsb = cls + (size_t)b * NP * C;
  const __bf16* img  = imgT + (size_t)b * HW * C;
  float cm = -3.0e38f, cs = 0.f;
  for (int nt = 0; nt < NP / 16; ++nt) {
    v8f acc = {};
    #pragma unroll
    for (int k = 0; k < 8; ++k) {
      v16bf a  = ld_frag(clsb, C, nt * 16, k * 32, lane);
      v16bf bb = ld_frag(img, C, p0, k * 32, lane);
      acc = wmma_bf16(a, bb, acc);
    }
    float tmax = -3.0e38f;
    #pragma unroll
    for (int v = 0; v < 8; ++v) {
      int n = nt * 16 + v + 8 * half;
      if (n < NQ) tmax = fmaxf(tmax, acc[v]);
    }
    tmax = fmaxf(tmax, __shfl_xor(tmax, 16, 32));
    float ts = 0.f;
    #pragma unroll
    for (int v = 0; v < 8; ++v) {
      int n = nt * 16 + v + 8 * half;
      if (n < NQ) ts += __expf(acc[v] - tmax);
    }
    ts += __shfl_xor(ts, 16, 32);
    float nm = fmaxf(cm, tmax);
    cs = cs * __expf(cm - nm) + ts * __expf(tmax - nm);
    cm = nm;
  }
  if (lane < 16) {
    cmax[b * HW + p0 + lane] = cm;
    csum[b * HW + p0 + lane] = cs;
  }
}

// ---------------- cls side: (softmax(attn,hw) @ img) @ Wc^T + cls ----------------
__global__ __launch_bounds__(256, 1) void k_cls(
    const __bf16* __restrict__ cls, const __bf16* __restrict__ imgT,
    const __bf16* __restrict__ imgC, const __bf16* __restrict__ wc,
    const float* __restrict__ rmax, const float* __restrict__ rsum,
    const float* __restrict__ me, float* __restrict__ out_cls) {
  int b  = blockIdx.x / (NP / 16);
  int n0 = (blockIdx.x % (NP / 16)) * 16;
  int wave = threadIdx.x >> 5, lane = threadIdx.x & 31, half = lane >> 4;
  __shared__ __bf16 Pbuf[16 * 256];
  __shared__ __bf16 Sbuf[16 * 256];
  const __bf16* clsb = cls + (size_t)b * NP * C;
  const __bf16* imT  = imgT + (size_t)b * HW * C;
  const __bf16* imC  = imgC + (size_t)b * C * HW;
  v16bf afr[8];
  #pragma unroll
  for (int k = 0; k < 8; ++k) afr[k] = ld_frag(clsb, C, n0, k * 32, lane);
  float rm[8], ri[8];
  #pragma unroll
  for (int v = 0; v < 8; ++v) {
    rm[v] = rmax[b * NP + n0 + v + 8 * half];
    ri[v] = 1.f / rsum[b * NP + n0 + v + 8 * half];
  }
  v8f accS0 = {}, accS1 = {};
  int j0 = wave * 32;
  for (int pc = 0; pc < HW / 256; ++pc) {
    int pbase = pc * 256;
    #pragma unroll
    for (int t = 0; t < 2; ++t) {
      int ptl = wave * 2 + t;
      v8f acc = {};
      #pragma unroll
      for (int k = 0; k < 8; ++k)
        acc = wmma_bf16(afr[k], ld_frag(imT, C, pbase + ptl * 16, k * 32, lane), acc);
      #pragma unroll
      for (int v = 0; v < 8; ++v) {
        float pv = __expf(acc[v] - rm[v]) * ri[v];
        Pbuf[(v + 8 * half) * 256 + ptl * 16 + (lane & 15)] = f2bf(pv);
      }
    }
    __syncthreads();
    #pragma unroll
    for (int kc = 0; kc < 8; ++kc) {
      v16bf a  = ld_frag(Pbuf, 256, 0, kc * 32, lane);
      v16bf b0 = ld_frag(imC, HW, j0, pbase + kc * 32, lane);
      v16bf b1 = ld_frag(imC, HW, j0 + 16, pbase + kc * 32, lane);
      accS0 = wmma_bf16(a, b0, accS0);
      accS1 = wmma_bf16(a, b1, accS1);
    }
    __syncthreads();
  }
  #pragma unroll
  for (int v = 0; v < 8; ++v) {
    Sbuf[(v + 8 * half) * 256 + j0 + (lane & 15)]      = f2bf(accS0[v]);
    Sbuf[(v + 8 * half) * 256 + j0 + 16 + (lane & 15)] = f2bf(accS1[v]);
  }
  __syncthreads();
  #pragma unroll
  for (int t = 0; t < 2; ++t) {
    int jo = wave * 32 + t * 16;
    v8f acc = {};
    #pragma unroll
    for (int kc = 0; kc < 8; ++kc) {
      v16bf a  = ld_frag(Sbuf, 256, 0, kc * 32, lane);
      v16bf bb = ld_frag(wc, C, jo, kc * 32, lane);
      acc = wmma_bf16(a, bb, acc);
    }
    int ch = jo + (lane & 15);
    #pragma unroll
    for (int v = 0; v < 8; ++v) {
      int n = n0 + v + 8 * half;
      if (n < NQ) {
        size_t o = ((size_t)(b * NQ + n)) * C + ch;
        out_cls[o] = me[o] + acc[v];
      }
    }
  }
}

// ---------------- feat side: (softmax(attn,N)^T @ cls) @ Wf^T + img ----------------
__global__ __launch_bounds__(256, 1) void k_feat(
    const __bf16* __restrict__ cls, const __bf16* __restrict__ clsT,
    const __bf16* __restrict__ imgT, const __bf16* __restrict__ wf,
    const float* __restrict__ cmax, const float* __restrict__ csum,
    const float* __restrict__ mf, float* __restrict__ out_img) {
  int b  = blockIdx.x / (HW / 32);
  int pbase = (blockIdx.x % (HW / 32)) * 32;
  int wave = threadIdx.x >> 5, lane = threadIdx.x & 31, half = lane >> 4;
  __shared__ __bf16 PT[32 * 160];   // [pixel][class]
  __shared__ __bf16 T1[32 * 256];   // [pixel][channel]
  const __bf16* clsb  = cls  + (size_t)b * NP * C;
  const __bf16* clsTb = clsT + (size_t)b * C * NP;
  const __bf16* imT   = imgT + (size_t)b * HW * C;
  // stage 1: attn + column softmax -> PT
  for (int idx = wave; idx < 20; idx += 8) {
    int nt = idx >> 1, pt = idx & 1;
    v8f acc = {};
    #pragma unroll
    for (int k = 0; k < 8; ++k) {
      v16bf a  = ld_frag(clsb, C, nt * 16, k * 32, lane);
      v16bf bb = ld_frag(imT, C, pbase + pt * 16, k * 32, lane);
      acc = wmma_bf16(a, bb, acc);
    }
    int p = pbase + pt * 16 + (lane & 15);
    float cm = cmax[b * HW + p];
    float ci = 1.f / csum[b * HW + p];
    #pragma unroll
    for (int v = 0; v < 8; ++v) {
      int n = nt * 16 + v + 8 * half;
      float pv = (n < NQ) ? __expf(acc[v] - cm) * ci : 0.f;
      PT[(pt * 16 + (lane & 15)) * 160 + n] = f2bf(pv);
    }
  }
  __syncthreads();
  // stage 2: T1 = PT @ cls   (K = 160 classes)
  int j0 = wave * 32;
  #pragma unroll
  for (int t = 0; t < 4; ++t) {
    int pt = t >> 1;
    int j  = j0 + (t & 1) * 16;
    v8f acc = {};
    #pragma unroll
    for (int kc = 0; kc < 5; ++kc) {
      v16bf a  = ld_frag(PT, 160, pt * 16, kc * 32, lane);
      v16bf bb = ld_frag(clsTb, NP, j, kc * 32, lane);
      acc = wmma_bf16(a, bb, acc);
    }
    #pragma unroll
    for (int v = 0; v < 8; ++v)
      T1[(pt * 16 + v + 8 * half) * 256 + j + (lane & 15)] = f2bf(acc[v]);
  }
  __syncthreads();
  // stage 3: out = T1 @ Wf^T + img  (epilogue fused, NCHW contiguous v8f I/O)
  #pragma unroll
  for (int t = 0; t < 4; ++t) {
    int pt = t >> 1;
    int j  = j0 + (t & 1) * 16;
    v8f acc = {};
    #pragma unroll
    for (int kc = 0; kc < 8; ++kc) {
      v16bf a  = ld_frag(T1, 256, pt * 16, kc * 32, lane);
      v16bf bb = ld_frag(wf, C, j, kc * 32, lane);
      acc = wmma_bf16(a, bb, acc);
    }
    int ch = j + (lane & 15);
    size_t o = ((size_t)(b * C + ch)) * HW + pbase + pt * 16 + 8 * half;
    v8f base = *(const v8f*)(mf + o);
    v8f res;
    #pragma unroll
    for (int v = 0; v < 8; ++v) res[v] = base[v] + acc[v];
    *(v8f*)(out_img + o) = res;
  }
}

extern "C" void kernel_launch(void* const* d_in, const int* in_sizes, int n_in,
                              void* d_out, int out_size, void* d_ws, size_t ws_size,
                              hipStream_t stream) {
  const float* me = (const float*)d_in[0];   // mask_embed   [8,150,256]
  const float* mf = (const float*)d_in[1];   // mask_features[8,256,128,128]
  const float* Wc = (const float*)d_in[2];   // W_cls  [256,256]
  const float* Wf = (const float*)d_in[3];   // W_feat [256,256]
  char* ws = (char*)d_ws;
  __bf16* imgT = (__bf16*)(ws);                     // 64 MB [b][hw][c]
  __bf16* imgC = (__bf16*)(ws + 67108864ull);       // 64 MB [b][c][hw]
  __bf16* cls  = (__bf16*)(ws + 134217728ull);      // 640 KB [b][160][c]
  __bf16* clsT = (__bf16*)(ws + 134873088ull);      // 640 KB [b][c][160]
  __bf16* wcb  = (__bf16*)(ws + 135528448ull);      // 128 KB
  __bf16* wfb  = (__bf16*)(ws + 135659520ull);      // 128 KB
  float*  rmax = (float*)(ws + 135790592ull);       // 5 KB
  float*  rsum = (float*)(ws + 135795712ull);       // 5 KB
  float*  cmax = (float*)(ws + 135800832ull);       // 512 KB
  float*  csum = (float*)(ws + 136325120ull);       // 512 KB
  float* out_cls = (float*)d_out;
  float* out_img = out_cls + (size_t)NB * NQ * C;

  k_prep_img<<<NB * (C / 32) * (HW / 32), 256, 0, stream>>>(mf, imgT, imgC);
  int nsm = NB * NP * C + C * C;
  k_prep_small<<<(nsm + 255) / 256, 256, 0, stream>>>(me, Wc, Wf, cls, clsT, wcb, wfb);
  k_row_stats<<<NB * (NP / 16), 256, 0, stream>>>(cls, imgT, rmax, rsum);
  k_col_stats<<<NB * (HW / 128), 256, 0, stream>>>(cls, imgT, cmax, csum);
  k_cls<<<NB * (NP / 16), 256, 0, stream>>>(cls, imgT, imgC, wcb, rmax, rsum, me, out_cls);
  k_feat<<<NB * (HW / 32), 256, 0, stream>>>(cls, clsT, imgT, wfb, cmax, csum, mf, out_img);
}
